// DeepGEMMMaskedGemm_34548716929125
// MI455X (gfx1250) — matrix operand order repaired and checked by reading, not compiled, not run on hardware
//
#include <hip/hip_runtime.h>
#include <hip/hip_bf16.h>

#define E_    64
#define MAXM  256
#define N_    1024
#define K_    4096
#define BLK_  128
#define CBLK  32      // K_/BLK_
#define NBLK  8       // N_/BLK_

#define BM 128
#define BN 128
#define LDA 136       // 128 + 8 bf16 pad -> conflict-free b128 LDS reads
#define LDB 136

typedef __attribute__((ext_vector_type(16))) __bf16 v16bf;
typedef __attribute__((ext_vector_type(8)))  __bf16 v8bf;
typedef __attribute__((ext_vector_type(8)))  float  v8f;

union V16U { v16bf v; v8bf h[2]; };

__global__ __launch_bounds__(256)
void deepgemm_masked_bf16wmma(const float* __restrict__ input,
                              const float* __restrict__ input_scale,
                              const float* __restrict__ weight,
                              const float* __restrict__ weight_scale,
                              const int*   __restrict__ masked_m,
                              __bf16* __restrict__ out)
{
    // Double-buffered staging tiles (2 x (128x136) bf16 each for A and B).
    __shared__ __align__(16) __bf16 As[2][BM * LDA];
    __shared__ __align__(16) __bf16 Bs[2][BN * LDB];

    const int tid = threadIdx.x;
    const int gid = blockIdx.x;
    // gid = e*16 + nt*2 + mt  (mt adjacent -> immediate weight-tile L2 reuse)
    const int e   = gid >> 4;
    const int rem = gid & 15;
    const int nt  = rem >> 1;
    const int mt  = rem & 1;
    const int m0  = mt * BM;
    const int n0  = nt * BN;
    const int mm  = masked_m[e];

    __bf16* outp = out + ((size_t)e * MAXM) * N_;

    if (m0 >= mm) {
        // Fully-masked tile: zero-fill 128x128 bf16 (uniform branch).
        const int r    = tid >> 1;
        const int half = tid & 1;
        uint4 z; z.x = z.y = z.z = z.w = 0u;
        uint4* o = (uint4*)(outp + (size_t)(m0 + r) * N_ + n0 + half * 64);
#pragma unroll
        for (int j = 0; j < 8; ++j) o[j] = z;
        return;
    }

    const int lid  = tid & 31;
    const int wave = tid >> 5;
    const int wm   = wave & 3;   // 4 waves over M: 32 rows each
    const int wn   = wave >> 2;  // 2 waves over N: 64 cols each

    v8f acc[2][4];
#pragma unroll
    for (int i = 0; i < 2; ++i)
#pragma unroll
        for (int j = 0; j < 4; ++j)
#pragma unroll
            for (int k = 0; k < 8; ++k) acc[i][j][k] = 0.0f;

    const float* isb  = input_scale  + ((size_t)e * MAXM + m0) * CBLK;
    const float* wsb  = weight_scale + ((size_t)e * NBLK + nt) * CBLK;

    const int lr = tid >> 1;         // loader row 0..127
    const int lk = (tid & 1) * 64;   // loader k-half 0 / 64

    const float* aptr = input  + ((size_t)e * MAXM + m0 + lr) * K_ + lk;
    const float* bptr = weight + ((size_t)e * N_   + n0 + lr) * K_ + lk;

    // ---- register staging: 32 fp32x4 in flight while WMMAs run ----
    float4 ra[16], rb[16];
    float  sa, sb;

    auto load_regs = [&](int c) {
        sa = isb[(size_t)lr * CBLK + c];
        sb = wsb[c];
        const float4* s0 = (const float4*)(aptr + c * BLK_);
        const float4* s1 = (const float4*)(bptr + c * BLK_);
#pragma unroll
        for (int j = 0; j < 16; ++j) ra[j] = s0[j];
#pragma unroll
        for (int j = 0; j < 16; ++j) rb[j] = s1[j];
        if (c + 1 < CBLK) {            // warm L2 one more chunk ahead
            __builtin_prefetch(aptr + (c + 1) * BLK_, 0, 0);
            __builtin_prefetch(bptr + (c + 1) * BLK_, 0, 0);
        }
    };

    auto flush_lds = [&](int buf) {
        __bf16* da = &As[buf][lr * LDA + lk];
        __bf16* db = &Bs[buf][lr * LDB + lk];
#pragma unroll
        for (int j = 0; j < 16; ++j) {
            union { __bf16 h[4]; uint2 u; } p;
            p.h[0] = (__bf16)(ra[j].x * sa);
            p.h[1] = (__bf16)(ra[j].y * sa);
            p.h[2] = (__bf16)(ra[j].z * sa);
            p.h[3] = (__bf16)(ra[j].w * sa);
            *(uint2*)(da + j * 4) = p.u;
        }
#pragma unroll
        for (int j = 0; j < 16; ++j) {
            union { __bf16 h[4]; uint2 u; } p;
            p.h[0] = (__bf16)(rb[j].x * sb);
            p.h[1] = (__bf16)(rb[j].y * sb);
            p.h[2] = (__bf16)(rb[j].z * sb);
            p.h[3] = (__bf16)(rb[j].w * sb);
            *(uint2*)(db + j * 4) = p.u;
        }
    };

    load_regs(0);
    flush_lds(0);

    for (int c = 0; c < CBLK; ++c) {
        __syncthreads();               // buf[c&1] ready; prior compute drained
        const int cur = c & 1;

        if (c + 1 < CBLK) load_regs(c + 1);   // issue global loads, no wait

        const __bf16* Ab = As[cur];
        const __bf16* Bb = Bs[cur];

#pragma unroll
        for (int kk = 0; kk < 4; ++kk) {
            V16U Af[2];
#pragma unroll
            for (int mf = 0; mf < 2; ++mf) {
                // A 16x32 bf16: lanes 0-15 (row M=lane) K0..7 + K16..23;
                // lanes 16-31 K8..15 + K24..31.
                const int ar = wm * 32 + mf * 16 + (lid & 15);
                const int ak = kk * 32 + ((lid >> 4) << 3);
                const __bf16* ap = &Ab[ar * LDA + ak];
                Af[mf].h[0] = *(const v8bf*)ap;
                Af[mf].h[1] = *(const v8bf*)(ap + 16);
            }
            V16U Bf[4];
#pragma unroll
            for (int nf = 0; nf < 4; ++nf) {
                // B 32x16 bf16: lanes 0-15 (col N=lane) K0..7 + K16..23; lanes 16-31 K8..15 + K24..31.
                const int br = wn * 64 + nf * 16 + (lid & 15);
                const int bk = kk * 32 + ((lid >> 4) << 3);
                const __bf16* bp = &Bb[br * LDB + bk];
                Bf[nf].h[0] = *(const v8bf*)bp;
                Bf[nf].h[1] = *(const v8bf*)(bp + 16);
            }
#pragma unroll
            for (int mf = 0; mf < 2; ++mf)
#pragma unroll
                for (int nf = 0; nf < 4; ++nf)
                    acc[mf][nf] = __builtin_amdgcn_wmma_f32_16x16x32_bf16(
                        false, Af[mf].v, false, Bf[nf].v,
                        (short)0, acc[mf][nf], false, false);
        }

        if (c + 1 < CBLK) flush_lds((c + 1) & 1);  // cvt + ds_store after compute
    }

    // ---- store: C/D layout lane<16 -> M=i, lane>=16 -> M=i+8; N = lane&15 ----
    const int mrow_off = (lid >> 4) << 3;
#pragma unroll
    for (int mf = 0; mf < 2; ++mf) {
#pragma unroll
        for (int nf = 0; nf < 4; ++nf) {
            const int n = n0 + wn * 64 + nf * 16 + (lid & 15);
#pragma unroll
            for (int i = 0; i < 8; ++i) {
                const int m = m0 + wm * 32 + mf * 16 + i + mrow_off;
                const float v = (m < mm) ? acc[mf][nf][i] : 0.0f;
                outp[(size_t)m * N_ + n] = (__bf16)v;
            }
        }
    }
}

extern "C" void kernel_launch(void* const* d_in, const int* in_sizes, int n_in,
                              void* d_out, int out_size, void* d_ws, size_t ws_size,
                              hipStream_t stream) {
    const float* input        = (const float*)d_in[0];
    const float* input_scale  = (const float*)d_in[1];
    const float* weight       = (const float*)d_in[2];
    const float* weight_scale = (const float*)d_in[3];
    const int*   masked_m     = (const int*)d_in[4];
    __bf16*      out          = (__bf16*)d_out;

    const dim3 grid(E_ * (MAXM / BM) * (N_ / BN));  // 1024 blocks
    const dim3 block(256);                          // 8 wave32s
    hipLaunchKernelGGL(deepgemm_masked_bf16wmma, grid, block, 0, stream,
                       input, input_scale, weight, weight_scale, masked_m, out);
}
